// AttentionPairBias_88897233093054
// MI455X (gfx1250) — matrix operand order repaired and checked by reading, not compiled, not run on hardware
//
#include <hip/hip_runtime.h>
#include <hip/hip_bf16.h>

#define HIDDEN   768
#define PAIRD    128
#define HEADS    12
#define HEAD_DIM 64
#define SEQ      512
#define BATCH    2
#define MROWS    (BATCH * SEQ)          // 1024
#define SCALE_F  0.125f                 // 64^-0.5
#define EPS_F    1e-5f

typedef __attribute__((ext_vector_type(16))) __bf16 v16bf;
typedef __attribute__((ext_vector_type(8)))  float  v8f;

union Frag { v16bf v; uint4 q[2]; };

__device__ __forceinline__ unsigned short f2bf(float f) {
    unsigned int u = __builtin_bit_cast(unsigned int, f);
    u += 0x7FFFu + ((u >> 16) & 1u);
    return (unsigned short)(u >> 16);
}

__device__ __forceinline__ float wave_sum32(float x) {
#pragma unroll
    for (int o = 16; o > 0; o >>= 1) x += __shfl_xor(x, o, 32);
    return x;
}

// --- CDNA5 async memory->LDS DMA (ASYNCcnt-tracked), per-lane 16B ----------
__device__ __forceinline__ unsigned lds_off32(const void* p) {
    // flat AS3->AS0 cast puts the LDS byte offset in the low 32 bits
    return (unsigned)(unsigned long long)p;
}
__device__ __forceinline__ void async_load_b128(unsigned lds_off, const void* gaddr) {
    asm volatile("global_load_async_to_lds_b128 %0, %1, off"
                 :: "v"(lds_off), "v"(gaddr) : "memory");
}
__device__ __forceinline__ void wait_async0() {
#if __has_builtin(__builtin_amdgcn_s_wait_asynccnt)
    __builtin_amdgcn_s_wait_asynccnt(0);
#else
    asm volatile("s_wait_asynccnt 0x0" ::: "memory");
#endif
}

// ---------------------------------------------------------------------------
// K1: LayerNorm(s) -> bf16   (row-per-block, 256 threads, 3 elems/thread)
// ---------------------------------------------------------------------------
__global__ __launch_bounds__(256) void ln_s_kernel(
    const float* __restrict__ s, const float* __restrict__ g,
    const float* __restrict__ bb, unsigned short* __restrict__ out)
{
    const int row = blockIdx.x;
    const int t   = threadIdx.x;
    const float x0 = s[row * HIDDEN + t];
    const float x1 = s[row * HIDDEN + t + 256];
    const float x2 = s[row * HIDDEN + t + 512];
    float s1 = x0 + x1 + x2;
    float s2 = x0 * x0 + x1 * x1 + x2 * x2;
    s1 = wave_sum32(s1);
    s2 = wave_sum32(s2);
    __shared__ float r1[8], r2[8];
    const int wid = t >> 5, lane = t & 31;
    if (lane == 0) { r1[wid] = s1; r2[wid] = s2; }
    __syncthreads();
    float S1 = 0.f, S2 = 0.f;
#pragma unroll
    for (int i = 0; i < 8; ++i) { S1 += r1[i]; S2 += r2[i]; }
    const float mean = S1 * (1.0f / HIDDEN);
    const float var  = S2 * (1.0f / HIDDEN) - mean * mean;
    const float inv  = rsqrtf(var + EPS_F);
    out[row * HIDDEN + t]       = f2bf((x0 - mean) * inv * g[t]       + bb[t]);
    out[row * HIDDEN + t + 256] = f2bf((x1 - mean) * inv * g[t + 256] + bb[t + 256]);
    out[row * HIDDEN + t + 512] = f2bf((x2 - mean) * inv * g[t + 512] + bb[t + 512]);
}

// ---------------------------------------------------------------------------
// K2/K5: WMMA GEMM  Y = X(bf16, MxK) @ W(f32, KxN) + bias, mode epilogues.
// 128 threads = 4 waves; block tile 64x64; wave tile 16x64; K step 32.
// X tile staged via global_load_async_to_lds_b128 (no VGPR round-trip);
// W tile needs f32->bf16 + transpose so it goes through VALU.
// MODE 0: QKV -> scatter q,k bf16 (B,H,N,D) and v transposed (B,H,D,N)
// MODE 1: gate = sigmoid(Y) f32
// MODE 2: out = gate * Y  (f32, final output)
// ---------------------------------------------------------------------------
template <int MODE>
__global__ __launch_bounds__(128) void gemm_wmma_kernel(
    const unsigned short* __restrict__ X, const float* __restrict__ W,
    const float* __restrict__ bias, int K, int Nout,
    unsigned short* __restrict__ q_ws, unsigned short* __restrict__ k_ws,
    unsigned short* __restrict__ vt_ws, float* __restrict__ gate_ws,
    const float* __restrict__ gate_in, float* __restrict__ out_f32)
{
    __shared__ unsigned short lx[64][40];   // X tile, [m][k], padded (16B-aligned rows)
    __shared__ unsigned short lw[64][40];   // W tile transposed, [n][k], padded

    const int m0  = blockIdx.y * 64;
    const int n0  = blockIdx.x * 64;
    const int tid = threadIdx.x;
    const int wid = tid >> 5, lane = tid & 31;
    const int lm  = lane & 15, kg = lane >> 4;

    v8f acc[4];
#pragma unroll
    for (int nt = 0; nt < 4; ++nt)
#pragma unroll
        for (int r = 0; r < 8; ++r) acc[nt][r] = 0.f;

    const int xrow = tid >> 1, xkoff = (tid & 1) * 16;
    const unsigned lx_dst0 = lds_off32(&lx[xrow][xkoff]);
    const unsigned lx_dst1 = lds_off32(&lx[xrow][xkoff + 8]);

    for (int k0 = 0; k0 < K; k0 += 32) {
        // X tile: 64 rows x 32 halfs, async DMA straight into LDS
        {
            const unsigned short* src = &X[(size_t)(m0 + xrow) * K + k0 + xkoff];
            async_load_b128(lx_dst0, src);
            async_load_b128(lx_dst1, src + 8);
        }
        // W tile: 32 k x 64 n f32 -> bf16 transposed into lw[n][k]
        {
            const int k = tid >> 2, nb = (tid & 3) * 16;
            const float* src = &W[(size_t)(k0 + k) * Nout + n0 + nb];
#pragma unroll
            for (int i = 0; i < 16; i += 4) {
                const float4 f = *(const float4*)&src[i];
                lw[nb + i + 0][k] = f2bf(f.x);
                lw[nb + i + 1][k] = f2bf(f.y);
                lw[nb + i + 2][k] = f2bf(f.z);
                lw[nb + i + 3][k] = f2bf(f.w);
            }
            if (k0 + 32 < K)
                __builtin_prefetch(&W[(size_t)(k0 + 32 + k) * Nout + n0 + nb], 0, 0);
        }
        wait_async0();
        __syncthreads();

        Frag a;
        a.q[0] = *(const uint4*)&lx[wid * 16 + lm][kg * 8];
        a.q[1] = *(const uint4*)&lx[wid * 16 + lm][16 + kg * 8];
#pragma unroll
        for (int nt = 0; nt < 4; ++nt) {
            Frag bf;
            bf.q[0] = *(const uint4*)&lw[nt * 16 + lm][kg * 8];
            bf.q[1] = *(const uint4*)&lw[nt * 16 + lm][16 + kg * 8];
            acc[nt] = __builtin_amdgcn_wmma_f32_16x16x32_bf16(
                false, a.v, false, bf.v, (short)0, acc[nt], false, false);
        }
        __syncthreads();
    }

    // epilogue
#pragma unroll
    for (int nt = 0; nt < 4; ++nt) {
#pragma unroll
        for (int r = 0; r < 8; ++r) {
            const int m   = m0 + wid * 16 + r + 8 * kg;
            const int col = n0 + nt * 16 + lm;
            const float y = acc[nt][r] + bias[col];
            if (MODE == 0) {
                const int qi = col / HIDDEN;
                const int rem = col - qi * HIDDEN;
                const int h = rem >> 6, d = rem & 63;
                const int b = m >> 9, rw = m & 511;
                if (qi == 0) {
                    q_ws[((((size_t)b * HEADS + h) * SEQ) + rw) * HEAD_DIM + d] = f2bf(y);
                } else if (qi == 1) {
                    k_ws[((((size_t)b * HEADS + h) * SEQ) + rw) * HEAD_DIM + d] = f2bf(y);
                } else {
                    vt_ws[((((size_t)b * HEADS + h) * HEAD_DIM) + d) * SEQ + rw] = f2bf(y);
                }
            } else if (MODE == 1) {
                gate_ws[(size_t)m * HIDDEN + col] = 1.0f / (1.0f + __expf(-y));
            } else {
                out_f32[(size_t)m * HIDDEN + col] =
                    gate_in[(size_t)m * HIDDEN + col] * y;
            }
        }
    }
}

// ---------------------------------------------------------------------------
// K3: fused LN(z) + (128 -> 12) projection, one wave32 per (b,i,j) row.
// Single pass over the 268MB z tensor => bandwidth-bound, ~11.5us floor.
// ---------------------------------------------------------------------------
__global__ __launch_bounds__(256) void pair_bias_kernel(
    const float* __restrict__ z, const float* __restrict__ g,
    const float* __restrict__ bb, const float* __restrict__ Wpb,
    float* __restrict__ out)
{
    const int wid  = threadIdx.x >> 5, lane = threadIdx.x & 31;
    const long idx = (long)blockIdx.x * 8 + wid;       // < BATCH*SEQ*SEQ
    const int b = (int)(idx >> 18);
    const int rem = (int)(idx & 262143);
    const int i = rem >> 9, j = rem & 511;

    const float4 x = *(const float4*)&z[(size_t)idx * PAIRD + lane * 4];
    float s1 = x.x + x.y + x.z + x.w;
    float s2 = x.x * x.x + x.y * x.y + x.z * x.z + x.w * x.w;
    s1 = wave_sum32(s1);
    s2 = wave_sum32(s2);
    const float mean = s1 * (1.0f / PAIRD);
    const float inv  = rsqrtf(s2 * (1.0f / PAIRD) - mean * mean + EPS_F);

    const float4 gg = *(const float4*)&g[lane * 4];
    const float4 bt = *(const float4*)&bb[lane * 4];
    const float n0 = (x.x - mean) * inv * gg.x + bt.x;
    const float n1 = (x.y - mean) * inv * gg.y + bt.y;
    const float n2 = (x.z - mean) * inv * gg.z + bt.z;
    const float n3 = (x.w - mean) * inv * gg.w + bt.w;

    const float* wp = &Wpb[(lane * 4) * HEADS];
    float ph[HEADS];
#pragma unroll
    for (int h = 0; h < HEADS; ++h)
        ph[h] = n0 * wp[h] + n1 * wp[HEADS + h] + n2 * wp[2 * HEADS + h] +
                n3 * wp[3 * HEADS + h];
#pragma unroll
    for (int h = 0; h < HEADS; ++h) ph[h] = wave_sum32(ph[h]);

    float v = 0.f;
#pragma unroll
    for (int h = 0; h < HEADS; ++h)
        if (lane == h) v = ph[h];
    if (lane < HEADS)
        out[((((size_t)b * HEADS + lane) * SEQ) + i) * SEQ + j] = v;
}

// ---------------------------------------------------------------------------
// K4: attention per (b,h,16-query tile). 4 waves.
//  stage 0: async-DMA the whole 16x512 f32 pair-bias slab into LDS sc[][]
//           (contiguous 2KB rows, 16 lane-wide b128 issues) + load q tile
//  stage 1: scores 16x512 via WMMA (K=64 -> 2 steps), sc = acc*SCALE + sc
//  stage 2: softmax (8 lanes per row), probs -> bf16 LDS
//  stage 3: P(16x512) @ V(512x64) via WMMA; V pre-transposed (B,H,D,N)
// ---------------------------------------------------------------------------
__global__ __launch_bounds__(128) void attn_kernel(
    const unsigned short* __restrict__ q_ws, const unsigned short* __restrict__ k_ws,
    const unsigned short* __restrict__ vt_ws, const float* __restrict__ pbias,
    unsigned short* __restrict__ attn_out)
{
    __shared__ unsigned short qs[16][64];
    __shared__ float sc[16][512];
    __shared__ unsigned short pb[16][520];
    __shared__ float rsum[16];

    const int bh    = blockIdx.x >> 5;            // b*HEADS + h
    const int qt    = blockIdx.x & 31;
    const int b     = bh / HEADS, h = bh % HEADS;
    const int qbase = qt * 16;
    const int tid   = threadIdx.x;
    const int wid   = tid >> 5, lane = tid & 31;
    const int lm    = lane & 15, kg = lane >> 4;

    // stage 0a: async bias slab -> sc (16 rows x 2KB; one b128 per lane per row)
    {
        const float* src = &pbias[(((size_t)bh * SEQ) + qbase) * SEQ + tid * 4];
#pragma unroll
        for (int r = 0; r < 16; ++r)
            async_load_b128(lds_off32(&sc[r][tid * 4]), src + (size_t)r * SEQ);
    }
    // stage 0b: q tile (16x64 bf16)
    {
        const int row = tid >> 3, c = (tid & 7) * 8;
        *(uint4*)&qs[row][c] =
            *(const uint4*)&q_ws[(((size_t)bh * SEQ) + qbase + row) * HEAD_DIM + c];
    }
    wait_async0();
    __syncthreads();

    // stage 1: scores (bias already sitting in sc)
    for (int kt = wid; kt < 32; kt += 4) {
        v8f acc;
#pragma unroll
        for (int r = 0; r < 8; ++r) acc[r] = 0.f;
#pragma unroll
        for (int dc = 0; dc < 2; ++dc) {
            Frag a, bf;
            a.q[0] = *(const uint4*)&qs[lm][dc * 32 + kg * 8];
            a.q[1] = *(const uint4*)&qs[lm][dc * 32 + 16 + kg * 8];
            const size_t kb =
                (((size_t)bh * SEQ) + kt * 16 + lm) * HEAD_DIM + dc * 32 + kg * 8;
            bf.q[0] = *(const uint4*)&k_ws[kb];
            bf.q[1] = *(const uint4*)&k_ws[kb + 16];
            acc = __builtin_amdgcn_wmma_f32_16x16x32_bf16(
                false, a.v, false, bf.v, (short)0, acc, false, false);
        }
#pragma unroll
        for (int r = 0; r < 8; ++r) {
            const int row = r + 8 * kg;
            const int col = kt * 16 + lm;
            sc[row][col] = acc[r] * SCALE_F + sc[row][col];
        }
    }
    __syncthreads();

    // stage 2: softmax, 8 lanes per row
    {
        const int row = tid >> 3, sub = tid & 7;
        float mx = -1e30f;
        for (int j = sub; j < SEQ; j += 8) mx = fmaxf(mx, sc[row][j]);
#pragma unroll
        for (int o = 4; o > 0; o >>= 1) mx = fmaxf(mx, __shfl_xor(mx, o, 32));
        float sm = 0.f;
        for (int j = sub; j < SEQ; j += 8) {
            const float e = __expf(sc[row][j] - mx);
            pb[row][j] = f2bf(e);
            sm += e;
        }
#pragma unroll
        for (int o = 4; o > 0; o >>= 1) sm += __shfl_xor(sm, o, 32);
        if (sub == 0) rsum[row] = 1.0f / sm;
    }
    __syncthreads();

    // stage 3: P @ V, each wave one 16x16 n-tile
    {
        const int n0 = wid * 16;
        v8f acc;
#pragma unroll
        for (int r = 0; r < 8; ++r) acc[r] = 0.f;
        for (int kt = 0; kt < 16; ++kt) {
            Frag a, bf;
            a.q[0] = *(const uint4*)&pb[lm][kt * 32 + kg * 8];
            a.q[1] = *(const uint4*)&pb[lm][kt * 32 + 16 + kg * 8];
            const size_t vb =
                (((size_t)bh * HEAD_DIM) + n0 + lm) * SEQ + kt * 32 + kg * 8;
            bf.q[0] = *(const uint4*)&vt_ws[vb];
            bf.q[1] = *(const uint4*)&vt_ws[vb + 16];
            acc = __builtin_amdgcn_wmma_f32_16x16x32_bf16(
                false, a.v, false, bf.v, (short)0, acc, false, false);
        }
#pragma unroll
        for (int r = 0; r < 8; ++r) {
            const int row = r + 8 * kg;
            const int col = n0 + lm;
            const float v = acc[r] * rsum[row];
            attn_out[(((size_t)b * SEQ) + qbase + row) * HIDDEN + h * HEAD_DIM + col] =
                f2bf(v);
        }
    }
}

// ---------------------------------------------------------------------------
extern "C" void kernel_launch(void* const* d_in, const int* in_sizes, int n_in,
                              void* d_out, int out_size, void* d_ws, size_t ws_size,
                              hipStream_t stream)
{
    const float* s      = (const float*)d_in[0];
    const float* z      = (const float*)d_in[1];
    const float* ln_s_g = (const float*)d_in[2];
    const float* ln_s_b = (const float*)d_in[3];
    const float* ln_z_g = (const float*)d_in[4];
    const float* ln_z_b = (const float*)d_in[5];
    const float* Wqkv   = (const float*)d_in[6];
    const float* bqkv   = (const float*)d_in[7];
    const float* Wg     = (const float*)d_in[8];
    const float* bg     = (const float*)d_in[9];
    const float* Wpb    = (const float*)d_in[10];
    const float* Wo     = (const float*)d_in[11];
    const float* bo     = (const float*)d_in[12];
    float* out = (float*)d_out;

    char* ws = (char*)d_ws;
    const size_t SLN   = (size_t)MROWS * HIDDEN * 2;                 // s_ln bf16
    const size_t SQK   = (size_t)BATCH * HEADS * SEQ * HEAD_DIM * 2; // bf16
    unsigned short* s_ln  = (unsigned short*)(ws);
    unsigned short* q_ws  = (unsigned short*)(ws + SLN);
    unsigned short* k_ws  = (unsigned short*)(ws + SLN + SQK);
    unsigned short* vt_ws = (unsigned short*)(ws + SLN + 2 * SQK);
    unsigned short* a_ws  = (unsigned short*)(ws + SLN + 3 * SQK);   // attn out bf16
    float* gate_ws        = (float*)(ws + SLN + 4 * SQK);
    float* pbias_ws       = (float*)(ws + SLN + 4 * SQK + (size_t)MROWS * HIDDEN * 4);

    // 1) LN(s) -> bf16
    ln_s_kernel<<<MROWS, 256, 0, stream>>>(s, ln_s_g, ln_s_b, s_ln);

    // 2) QKV GEMM (1024x768 @ 768x2304) + scatter q,k,v^T
    gemm_wmma_kernel<0><<<dim3(3 * HIDDEN / 64, MROWS / 64), 128, 0, stream>>>(
        s_ln, Wqkv, bqkv, HIDDEN, 3 * HIDDEN, q_ws, k_ws, vt_ws, nullptr, nullptr,
        nullptr);

    // 3) gate GEMM + sigmoid
    gemm_wmma_kernel<1><<<dim3(HIDDEN / 64, MROWS / 64), 128, 0, stream>>>(
        s_ln, Wg, bg, HIDDEN, HIDDEN, nullptr, nullptr, nullptr, gate_ws, nullptr,
        nullptr);

    // 4) fused LN(z) + pair-bias projection (single pass over 268MB)
    pair_bias_kernel<<<(BATCH * SEQ * SEQ) / 8, 256, 0, stream>>>(
        z, ln_z_g, ln_z_b, Wpb, pbias_ws);

    // 5) attention
    attn_kernel<<<BATCH * HEADS * (SEQ / 16), 128, 0, stream>>>(
        q_ws, k_ws, vt_ws, pbias_ws, a_ws);

    // 6) output GEMM + bias + gate
    gemm_wmma_kernel<2><<<dim3(HIDDEN / 64, MROWS / 64), 128, 0, stream>>>(
        a_ws, Wo, bo, HIDDEN, HIDDEN, nullptr, nullptr, nullptr, nullptr, gate_ws,
        out);
}
